// PathBasedModel_89764816487071
// MI455X (gfx1250) — compile-verified, matrix operand order checked by reading, not run on hardware
//
#include <hip/hip_runtime.h>
#include <hip/hip_bf16.h>
#include <stdint.h>

typedef __bf16 bf16_t;
typedef bf16_t       v16bf __attribute__((ext_vector_type(16)));
typedef float        v8f   __attribute__((ext_vector_type(8)));
typedef unsigned int u32x4 __attribute__((ext_vector_type(4)));
typedef int          i32x4 __attribute__((ext_vector_type(4)));
typedef int          i32x8 __attribute__((ext_vector_type(8)));

#if defined(__has_builtin)
#  if __has_builtin(__builtin_amdgcn_tensor_load_to_lds)
#    define HAVE_TDM 1
#  endif
#endif
#ifndef HAVE_TDM
#  define HAVE_TDM 0
#endif
#if __has_include(<hip/amd_detail/amd_gfx1250_TDM.h>)
#  define TDM_SIX_ARGS 1
#else
#  define TDM_SIX_ARGS 0
#endif

// ---------------------------------------------------------------------------
// float -> bf16 (round-to-nearest-even), stored as raw ushort
// ---------------------------------------------------------------------------
static __device__ __forceinline__ unsigned short f2bf(float f) {
    unsigned int u = __float_as_uint(f);
    unsigned int r = u + 0x7FFFu + ((u >> 16) & 1u);
    return (unsigned short)(r >> 16);
}

// Branch-free activations (TRANS v_exp co-executes with XDL WMMA; no saveexec)
static __device__ __forceinline__ float fast_sigmoid(float x) {
    x = fminf(fmaxf(x, -30.f), 30.f);
    return __fdividef(1.f, 1.f + __expf(-x));
}
static __device__ __forceinline__ float fast_tanh(float x) {
    x = fminf(fmaxf(x, -15.f), 15.f);
    float e = __expf(2.f * x);
    return __fdividef(e - 1.f, e + 1.f);
}

// ---------------------------------------------------------------------------
// Kernel 1: bf16 conversion.
//  - W/U (both dirs) transposed 128x512 -> 512x128 for WMMA B lanes
//  - proj transposed 384x128 -> 128x384
//  - path_element_emb straight copy-converted (halves gather traffic)
// ---------------------------------------------------------------------------
__global__ void k_convert(const float* __restrict__ Wf, const float* __restrict__ Uf,
                          const float* __restrict__ Wb, const float* __restrict__ Ub,
                          const float* __restrict__ Pm, const float* __restrict__ emb,
                          unsigned short* __restrict__ ws, int VNT)
{
    int id = blockIdx.x * blockDim.x + threadIdx.x;
    unsigned short* Wt_f = ws;
    unsigned short* Ut_f = ws + 65536;
    unsigned short* Wt_b = ws + 131072;
    unsigned short* Ut_b = ws + 196608;
    unsigned short* Pt   = ws + 262144;
    unsigned short* embT = ws + 311296;
    if (id < 65536) {
        int n = id >> 7, k = id & 127;
        Wt_f[(n << 7) + k] = f2bf(Wf[(k << 9) + n]);
    } else if (id < 131072) {
        int r = id - 65536; int n = r >> 7, k = r & 127;
        Ut_f[(n << 7) + k] = f2bf(Uf[(k << 9) + n]);
    } else if (id < 196608) {
        int r = id - 131072; int n = r >> 7, k = r & 127;
        Wt_b[(n << 7) + k] = f2bf(Wb[(k << 9) + n]);
    } else if (id < 262144) {
        int r = id - 196608; int n = r >> 7, k = r & 127;
        Ut_b[(n << 7) + k] = f2bf(Ub[(k << 9) + n]);
    } else if (id < 311296) {
        int r = id - 262144; int n = r / 384, k = r % 384;
        Pt[n * 384 + k] = f2bf(Pm[k * 128 + n]);
    } else {
        int r = id - 311296;
        if (r < VNT * 128) embT[r] = f2bf(emb[r]);
    }
}

// ---------------------------------------------------------------------------
// Kernel 2: gather leaf embeddings -> feat[:, 0:128) as bf16
// ---------------------------------------------------------------------------
__global__ void k_leaf(const float* __restrict__ leafE, const int* __restrict__ leaf_idx,
                       unsigned short* __restrict__ feat, int P)
{
    int id = blockIdx.x * blockDim.x + threadIdx.x;
    if (id >= P * 16) return;
    int p    = id >> 4;
    int cb   = (id & 15) * 8;
    int half = cb >> 6;
    int idx  = leaf_idx[p * 2 + half];
    const float*    src = leafE + (size_t)idx * 64 + (cb & 63);
    unsigned short* dst = feat + (size_t)p * 384 + cb;
#pragma unroll
    for (int j = 0; j < 8; ++j) dst[j] = f2bf(src[j]);
}

// ---------------------------------------------------------------------------
// Kernel 3: fused gather + masked LSTM (one direction per launch).
// Block = 256 threads (8 waves), 16 paths per block. Double-buffered x-tile,
// fetched one step ahead by TDM gather-mode (16 rows x 256B, 16-bit indices).
// Wave w owns gate-columns [w*16, w*16+16): 4 gate tiles, each accumulated by
// 8x V_WMMA_F32_16X16X32_BF16. W/U fragments are loop-invariant -> the
// compiler keeps them register-resident across the T loop.
// ---------------------------------------------------------------------------
__global__ __launch_bounds__(256) void k_lstm(
    const unsigned short* __restrict__ embT,       // [V_NT][128] bf16
    const int*   __restrict__ path_elems,          // [P][T]
    const int*   __restrict__ path_len,            // [P]
    const unsigned short* __restrict__ Wt,         // [512][128] bf16 (transposed)
    const unsigned short* __restrict__ Ut,         // [512][128] bf16 (transposed)
    const float* __restrict__ bias,                // [512]
    unsigned short* __restrict__ feat,             // [P][384] bf16
    int P, int T, int dir, int VNT)
{
    __shared__ alignas(16) unsigned short xs[2][16 * 128];   // x tiles, bf16
    __shared__ alignas(16) unsigned short hs[16 * 128];      // h tile, bf16
    __shared__ int plen[16];

    const int tid   = threadIdx.x;
    const int lane  = tid & 31;
    const int wave  = tid >> 5;
    const int pbase = blockIdx.x * 16;
    const int n0    = wave * 16;
    const int ncol  = n0 + (lane & 15);
    const int khalf = (lane >> 4) * 16;

    if (tid < 16) plen[tid] = path_len[pbase + tid];
    for (int i = tid; i < 16 * 128; i += 256) hs[i] = 0;

    v8f c_frag = {};
    v8f h_frag = {};
    float bgate[4];
#pragma unroll
    for (int g = 0; g < 4; ++g) bgate[g] = bias[g * 128 + ncol];

    const int row = tid >> 4;            // used by fallback gather
    const int cb  = (tid & 15) * 8;

#if HAVE_TDM
    // Build/issue one TDM gather-mode load: 16 rows of 128 bf16 into LDS tile.
    auto issue_gather = [&](unsigned short* dst, int tt) {
        u32x4 g0; i32x8 g1; i32x4 g2, g3;
        uint64_t ga = (uint64_t)(uintptr_t)embT;
        g0[0] = 0x80000001u;                               // count=1 | gather_mode
        g0[1] = (unsigned)(uintptr_t)dst;                  // lds_addr
        g0[2] = (unsigned)ga;                              // global_addr lo
        g0[3] = (unsigned)(ga >> 32) | 0x80000000u;        // addr hi | type=2
        g1[0] = 1 << 16;                                   // data_size = 2 bytes
        g1[1] = 128 << 16;                                 // tensor_dim0 = 128
        g1[2] = (VNT & 0xFFFF) << 16;                      // tensor_dim1 lo
        g1[3] = ((VNT >> 16) & 0xFFFF) | (128 << 16);      // dim1 hi | tile_dim0
        g1[4] = 16;                                        // tile_dim1 = 16 indices
        g1[5] = 128;                                       // tensor_dim0_stride
        g1[6] = 0;
        g1[7] = 0;
        int iv[16];
#pragma unroll
        for (int m = 0; m < 16; ++m) iv[m] = path_elems[(pbase + m) * T + tt];
#pragma unroll
        for (int j = 0; j < 4; ++j) g2[j] = (iv[2 * j] & 0xFFFF) | (iv[2 * j + 1] << 16);
#pragma unroll
        for (int j = 0; j < 4; ++j) g3[j] = (iv[8 + 2 * j] & 0xFFFF) | (iv[9 + 2 * j] << 16);
#if TDM_SIX_ARGS
        i32x8 zz = {};
        __builtin_amdgcn_tensor_load_to_lds(g0, g1, g2, g3, zz, 0);
#else
        __builtin_amdgcn_tensor_load_to_lds(g0, g1, g2, g3, 0);
#endif
    };
    if (wave == 0) issue_gather(xs[0], dir ? (T - 1) : 0);
#else
    {
        const int t0  = dir ? (T - 1) : 0;
        const int idx = path_elems[(pbase + row) * T + t0];
        *(uint4*)&xs[0][row * 128 + cb] = *(const uint4*)&embT[(size_t)idx * 128 + cb];
    }
#endif

#pragma unroll 1
    for (int s = 0; s < T; ++s) {
        const int t  = dir ? (T - 1 - s) : s;
        const int tn = dir ? (t - 1) : (t + 1);

        // prefetch next step's x-tile into the other buffer
#if HAVE_TDM
        if (wave == 0) {
            if (s + 1 < T) {
                issue_gather(xs[(s + 1) & 1], tn);
                __builtin_amdgcn_s_wait_tensorcnt(1);   // oldest (current tile) done
            } else {
                __builtin_amdgcn_s_wait_tensorcnt(0);
            }
        }
#else
        if (s + 1 < T) {
            const int idx = path_elems[(pbase + row) * T + tn];
            *(uint4*)&xs[(s + 1) & 1][row * 128 + cb] =
                *(const uint4*)&embT[(size_t)idx * 128 + cb];
        }
#endif
        __syncthreads();   // x-tile ready; previous step's h published

        const unsigned short* xbuf = xs[s & 1];

        // ---- z[g] = x@W + h@U via WMMA (fp32 accum) ----
        v8f z[4];
#pragma unroll
        for (int g = 0; g < 4; ++g) {
            v8f acc = {};
            const unsigned short* Wg = Wt + (size_t)(g * 128 + ncol) * 128;
            const unsigned short* Ug = Ut + (size_t)(g * 128 + ncol) * 128;
#pragma unroll
            for (int kc = 0; kc < 4; ++kc) {
                v16bf a = *(const v16bf*)&xbuf[(lane & 15) * 128 + kc * 32 + khalf];
                v16bf b = *(const v16bf*)&Wg[kc * 32 + khalf];
                acc = __builtin_amdgcn_wmma_f32_16x16x32_bf16(false, a, false, b,
                                                              (short)0, acc, false, false);
            }
#pragma unroll
            for (int kc = 0; kc < 4; ++kc) {
                v16bf a = *(const v16bf*)&hs[(lane & 15) * 128 + kc * 32 + khalf];
                v16bf b = *(const v16bf*)&Ug[kc * 32 + khalf];
                acc = __builtin_amdgcn_wmma_f32_16x16x32_bf16(false, a, false, b,
                                                              (short)0, acc, false, false);
            }
            z[g] = acc;
        }
        __syncthreads();   // hs fully consumed by all waves

        // ---- gates + masked state update; publish new h to LDS ----
#pragma unroll
        for (int v = 0; v < 8; ++v) {
            const int  m   = v + ((lane >> 4) << 3);
            const bool act = (t < plen[m]);
            float iv = fast_sigmoid(z[0][v] + bgate[0]);
            float fv = fast_sigmoid(z[1][v] + bgate[1]);
            float gv = fast_tanh   (z[2][v] + bgate[2]);
            float ov = fast_sigmoid(z[3][v] + bgate[3]);
            float cn = fv * c_frag[v] + iv * gv;
            float hn = ov * fast_tanh(cn);
            c_frag[v] = act ? cn : c_frag[v];
            h_frag[v] = act ? hn : h_frag[v];
            hs[m * 128 + ncol] = f2bf(h_frag[v]);
        }
    }
    __syncthreads();

    // final hidden state -> feat[:, 128 + dir*128 + ncol]
#pragma unroll
    for (int v = 0; v < 8; ++v) {
        const int m = v + ((lane >> 4) << 3);
        feat[(size_t)(pbase + m) * 384 + 128 + dir * 128 + ncol] = f2bf(h_frag[v]);
    }
}

// ---------------------------------------------------------------------------
// Kernel 4: projection GEMM (feat[P][384] @ proj^T -> full[P][128]) + scores.
// ---------------------------------------------------------------------------
__global__ __launch_bounds__(256) void k_proj(
    const unsigned short* __restrict__ feat,   // [P][384] bf16
    const unsigned short* __restrict__ Pt,     // [128][384] bf16 (transposed)
    const float* __restrict__ att,             // [128]
    float* __restrict__ full,                  // [P][128]
    float* __restrict__ scores,                // [P]
    int P)
{
    __shared__ alignas(16) unsigned short fs[16 * 384];
    __shared__ float ssc[16];
    const int tid   = threadIdx.x;
    const int lane  = tid & 31;
    const int wave  = tid >> 5;
    const int pbase = blockIdx.x * 16;
    const int n0    = wave * 16;
    const int ncol  = n0 + (lane & 15);
    const int khalf = (lane >> 4) * 16;

    for (int i = tid; i < 768; i += 256)
        ((uint4*)fs)[i] = ((const uint4*)(feat + (size_t)pbase * 384))[i];
    if (tid < 16) ssc[tid] = 0.f;
    __syncthreads();

    v8f acc = {};
    const unsigned short* Bp = Pt + (size_t)ncol * 384;
#pragma unroll
    for (int kc = 0; kc < 12; ++kc) {
        v16bf a = *(const v16bf*)&fs[(lane & 15) * 384 + kc * 32 + khalf];
        v16bf b = *(const v16bf*)&Bp[kc * 32 + khalf];
        acc = __builtin_amdgcn_wmma_f32_16x16x32_bf16(false, a, false, b,
                                                      (short)0, acc, false, false);
    }

    const float an = att[ncol];
#pragma unroll
    for (int v = 0; v < 8; ++v) {
        const int m = v + ((lane >> 4) << 3);
        full[(size_t)(pbase + m) * 128 + ncol] = acc[v];
        float sp = acc[v] * an;
#pragma unroll
        for (int o = 1; o < 16; o <<= 1) sp += __shfl_xor(sp, o, 32);
        if ((lane & 15) == 0) atomicAdd(&ssc[m], sp);
    }
    __syncthreads();
    if (tid < 16) scores[pbase + tid] = ssc[tid];
}

// ---------------------------------------------------------------------------
// Kernel 5: segment softmax + weighted feature sum (one block per sample).
// ---------------------------------------------------------------------------
__global__ __launch_bounds__(128) void k_segsoftmax(
    const float* __restrict__ scores, const float* __restrict__ full,
    const int* __restrict__ seg, float* __restrict__ out, int P)
{
    const int s   = blockIdx.x;
    const int tid = threadIdx.x;
    __shared__ float red[128];
    __shared__ float sh_mx, sh_den;

    int lo = 0, hi = P;
    while (lo < hi) { int mid = (lo + hi) >> 1; if (seg[mid] <  s) lo = mid + 1; else hi = mid; }
    const int start = lo;
    hi = P;
    while (lo < hi) { int mid = (lo + hi) >> 1; if (seg[mid] <= s) lo = mid + 1; else hi = mid; }
    const int end = lo;

    float mx = -3.402823466e38f;
    for (int i = start + tid; i < end; i += 128) mx = fmaxf(mx, scores[i]);
    red[tid] = mx; __syncthreads();
    for (int o = 64; o > 0; o >>= 1) { if (tid < o) red[tid] = fmaxf(red[tid], red[tid + o]); __syncthreads(); }
    if (tid == 0) sh_mx = red[0];
    __syncthreads();
    mx = sh_mx;

    float sm = 0.f;
    for (int i = start + tid; i < end; i += 128) sm += __expf(scores[i] - mx);
    red[tid] = sm; __syncthreads();
    for (int o = 64; o > 0; o >>= 1) { if (tid < o) red[tid] += red[tid + o]; __syncthreads(); }
    if (tid == 0) sh_den = red[0];
    __syncthreads();
    const float den = sh_den;

    float acc = 0.f;
    if (end > start && den > 0.f) {
        for (int p = start; p < end; ++p) {
            float w = __expf(scores[p] - mx) / den;
            acc += w * full[(size_t)p * 128 + tid];
        }
    }
    out[(size_t)s * 128 + tid] = acc;
}

// ---------------------------------------------------------------------------
// Host-side launch
// ---------------------------------------------------------------------------
extern "C" void kernel_launch(void* const* d_in, const int* in_sizes, int n_in,
                              void* d_out, int out_size, void* d_ws, size_t ws_size,
                              hipStream_t stream)
{
    const float* leafE      = (const float*)d_in[0];
    const float* elemE      = (const float*)d_in[1];
    const float* Wf         = (const float*)d_in[2];
    const float* Uf         = (const float*)d_in[3];
    const float* bf         = (const float*)d_in[4];
    const float* Wb         = (const float*)d_in[5];
    const float* Ub         = (const float*)d_in[6];
    const float* bb         = (const float*)d_in[7];
    const float* Pm         = (const float*)d_in[8];
    const float* att        = (const float*)d_in[9];
    const int*   path_elems = (const int*)d_in[10];
    const int*   path_len   = (const int*)d_in[11];
    const int*   leaf_idx   = (const int*)d_in[12];
    const int*   seg        = (const int*)d_in[13];

    const int P   = in_sizes[11];
    const int T   = in_sizes[10] / P;
    const int S   = out_size / 128;
    const int VNT = in_sizes[1] / 128;

    // workspace layout (ushort units)
    unsigned short* ws_u16 = (unsigned short*)d_ws;
    unsigned short* Wt_f = ws_u16;                               // 512*128
    unsigned short* Ut_f = ws_u16 + 65536;
    unsigned short* Wt_b = ws_u16 + 131072;
    unsigned short* Ut_b = ws_u16 + 196608;
    unsigned short* Pt   = ws_u16 + 262144;                      // 128*384
    unsigned short* embT = ws_u16 + 311296;                      // VNT*128 bf16
    unsigned short* feat = embT + (size_t)VNT * 128;             // P*384 bf16
    float* fullp  = (float*)(feat + (size_t)P * 384);            // P*128 f32
    float* scores = fullp + (size_t)P * 128;                     // P f32

    const int convN = 311296 + VNT * 128;
    k_convert<<<(convN + 255) / 256, 256, 0, stream>>>(Wf, Uf, Wb, Ub, Pm, elemE, ws_u16, VNT);
    k_leaf<<<(P * 16 + 255) / 256, 256, 0, stream>>>(leafE, leaf_idx, feat, P);
    k_lstm<<<P / 16, 256, 0, stream>>>(embT, path_elems, path_len, Wt_f, Ut_f, bf, feat, P, T, 0, VNT);
    k_lstm<<<P / 16, 256, 0, stream>>>(embT, path_elems, path_len, Wt_b, Ut_b, bb, feat, P, T, 1, VNT);
    k_proj<<<P / 16, 256, 0, stream>>>(feat, Pt, att, fullp, scores, P);
    k_segsoftmax<<<S, 128, 0, stream>>>(scores, fullp, seg, (float*)d_out, P);
}
